// Qwen3Experts_17849884082535
// MI455X (gfx1250) — matrix operand order repaired and checked by reading, not compiled, not run on hardware
//
#include <hip/hip_runtime.h>

// ---------------------------------------------------------------------------
// Qwen3 MoE experts (T=2048,H=2048,I=768,E=64,K=8) for gfx1250 (MI455X).
// bf16 WMMA grouped GEMMs, fp32 accumulate, expert-grouped token dispatch.
// Double-buffered LDS tiles, async global->LDS copies for A (ASYNCcnt),
// packed b32 transpose stores for B, v_rcp silu, L2 prefetch on weights.
// ---------------------------------------------------------------------------

typedef __attribute__((ext_vector_type(16))) __bf16 v16bf;
typedef __attribute__((ext_vector_type(8)))  float  v8f;

constexpr int T_   = 2048;
constexpr int H_   = 2048;
constexpr int I_   = 768;
constexpr int E_   = 64;
constexpr int K_   = 8;

constexpr int MT = 64;    // M tile (rows of expert group)
constexpr int NT = 128;   // N tile
constexpr int KT = 32;    // K step (bf16 wmma k)
constexpr int LDP = 40;   // LDS row pitch in half-words (32 + 8 pad, keeps 16B align)

constexpr int MAX_TILES = (T_ * K_) / MT + E_;   // 320

union FragBf { v16bf v; uint4 q[2]; };

__device__ __forceinline__ unsigned short f2bf(float f) {
    unsigned int u = __float_as_uint(f);
    u += 0x7FFFu + ((u >> 16) & 1u);          // round-to-nearest-even
    return (unsigned short)(u >> 16);
}
__device__ __forceinline__ unsigned int pack2bf(float a, float b) {
    return (unsigned int)f2bf(a) | ((unsigned int)f2bf(b) << 16);
}

// async global->LDS 16B copy (CDNA5 TDM-adjacent path, tracked by ASYNCcnt)
__device__ __forceinline__ void async_copy_b128(void* lds_dst, const void* gsrc) {
    unsigned lds = (unsigned)(size_t)lds_dst;     // aperture rule: LDS off = addr[31:0]
    asm volatile("global_load_async_to_lds_b128 %0, %1, off"
                 :: "v"(lds), "v"(gsrc) : "memory");
}
__device__ __forceinline__ void wait_async_all() {
    asm volatile("s_wait_asynccnt 0" ::: "memory");
}

// ---------------------------------------------------------------------------
// 1) zero output + counters (harness poisons buffers; we own init)
// ---------------------------------------------------------------------------
__global__ void init_kernel(float* __restrict__ out, int* __restrict__ counts,
                            int* __restrict__ fill, int* __restrict__ total) {
    int i = blockIdx.x * blockDim.x + threadIdx.x;            // i < T*H/4
    float4 z; z.x = z.y = z.z = z.w = 0.f;
    ((float4*)out)[i] = z;
    if (i < E_) { counts[i] = 0; fill[i] = 0; }
    if (i == 0) { *total = 0; }
}

// ---------------------------------------------------------------------------
// 2) top-8 routing + softmax + per-expert counts
// ---------------------------------------------------------------------------
__global__ void routing_kernel(const float* __restrict__ logits,
                               int* __restrict__ experts_sel,
                               float* __restrict__ wts_sel,
                               int* __restrict__ counts) {
    int t = blockIdx.x * blockDim.x + threadIdx.x;
    if (t >= T_) return;
    const float* row = logits + (size_t)t * E_;
    unsigned long long taken = 0ull;
    float vals[K_]; int ids[K_];
#pragma unroll
    for (int k = 0; k < K_; ++k) {
        float best = -__builtin_inff(); int bi = 0;
        for (int e = 0; e < E_; ++e) {
            if ((taken >> e) & 1ull) continue;
            float v = row[e];
            if (v > best) { best = v; bi = e; }
        }
        taken |= 1ull << bi;
        vals[k] = best; ids[k] = bi;
    }
    float m = vals[0], s = 0.f, p[K_];
#pragma unroll
    for (int k = 0; k < K_; ++k) { p[k] = __expf(vals[k] - m); s += p[k]; }
    float inv = __builtin_amdgcn_rcpf(s);
#pragma unroll
    for (int k = 0; k < K_; ++k) {
        experts_sel[t * K_ + k] = ids[k];
        wts_sel[t * K_ + k]     = p[k] * inv;
        atomicAdd(&counts[ids[k]], 1);
    }
}

// ---------------------------------------------------------------------------
// 3) prefix sum over expert counts + (expert, m-tile) worklist
// ---------------------------------------------------------------------------
__global__ void scan_kernel(const int* __restrict__ counts, int* __restrict__ offsets,
                            int* __restrict__ worklist, int* __restrict__ total) {
    if (threadIdx.x != 0 || blockIdx.x != 0) return;
    int off = 0, w = 0;
    for (int e = 0; e < E_; ++e) {
        offsets[e] = off;
        int c  = counts[e];
        int nt = (c + MT - 1) / MT;
        for (int j = 0; j < nt; ++j) worklist[w++] = (e << 16) | j;
        off += c;
    }
    offsets[E_] = off;
    *total = w;
}

// ---------------------------------------------------------------------------
// 4) counting-sort scatter: token id + routing weight into grouped order
// ---------------------------------------------------------------------------
__global__ void scatter_kernel(const int* __restrict__ experts_sel,
                               const float* __restrict__ wts_sel,
                               const int* __restrict__ offsets, int* __restrict__ fill,
                               int* __restrict__ token_perm, float* __restrict__ weight_perm) {
    int i = blockIdx.x * blockDim.x + threadIdx.x;
    if (i >= T_ * K_) return;
    int e   = experts_sel[i];
    int pos = offsets[e] + atomicAdd(&fill[e], 1);
    token_perm[pos]  = i >> 3;          // token id
    weight_perm[pos] = wts_sel[i];
}

// ---------------------------------------------------------------------------
// 5) hidden_states fp32 -> bf16 (done once, reused by all experts)
// ---------------------------------------------------------------------------
__global__ void convert_kernel(const float* __restrict__ hs, unsigned short* __restrict__ hsb) {
    int i = blockIdx.x * blockDim.x + threadIdx.x;            // i < T*H/4
    float4 v = ((const float4*)hs)[i];
    ((uint2*)hsb)[i] = make_uint2(pack2bf(v.x, v.y), pack2bf(v.z, v.w));
}

// ---------------------------------------------------------------------------
// 6) gemm1: act = silu(hs @ gate_w[e]) * (hs @ up_w[e])   [K-dim = H]
//    8 waves: 4(M) x 2(N); each wave 16x64 strip -> 8 wmma / k-step.
// ---------------------------------------------------------------------------
__global__ __launch_bounds__(256) void gemm1_kernel(
    const unsigned short* __restrict__ hs_bf,
    const float* __restrict__ gate_w, const float* __restrict__ up_w,
    const int* __restrict__ token_perm, const int* __restrict__ offsets,
    const int* __restrict__ worklist, const int* __restrict__ total_tiles,
    unsigned short* __restrict__ act) {

    int tile = blockIdx.y;
    if (tile >= *total_tiles) return;
    int wle = worklist[tile];
    int e = wle >> 16, mt = wle & 0xFFFF;
    int gstart = offsets[e], gend = offsets[e + 1];
    int mbase  = gstart + mt * MT;
    int mcount = gend - mbase; if (mcount > MT) mcount = MT;
    int nbase  = blockIdx.x * NT;

    __shared__ __align__(16) unsigned short As[2][MT][LDP];
    __shared__ __align__(16) unsigned short Bg[2][NT][LDP];   // transposed [n][k]
    __shared__ __align__(16) unsigned short Bu[2][NT][LDP];
    __shared__ int s_tok[MT];

    int tid = threadIdx.x;
    if (tid < MT) {
        int gr = mbase + tid;
        s_tok[tid] = (tid < mcount) ? token_perm[gr] : token_perm[mbase];
    }
    __syncthreads();

    int wid = tid >> 5, lane = tid & 31;
    int laneHi = lane >> 4, lm = lane & 15;
    int m_off = (wid & 3) * 16;
    int n_off = (wid >> 2) * 64;

    // per-thread load coordinates
    const int a_m = tid >> 2, a_q = tid & 3;                  // A: 1x16B per thread
    const size_t wbase = (size_t)e * H_ * I_;

    v8f accG[4], accU[4];
#pragma unroll
    for (int s = 0; s < 4; ++s) {
        accG[s] = (v8f){0.f,0.f,0.f,0.f,0.f,0.f,0.f,0.f};
        accU[s] = (v8f){0.f,0.f,0.f,0.f,0.f,0.f,0.f,0.f};
    }

    constexpr int NSTEPS = H_ / KT;                           // 64
#define G1_LOAD(BUF, KK0)                                                        \
    {                                                                            \
        int k0l = (KK0);                                                         \
        async_copy_b128(&As[BUF][a_m][a_q * 8],                                  \
            (const uint4*)(hs_bf + (size_t)s_tok[a_m] * H_ + k0l) + a_q);        \
        _Pragma("unroll")                                                        \
        for (int it = 0; it < 2; ++it) {                                         \
            int f  = it * 256 + tid;              /* 0..511 */                   \
            int kk = (f >> 5) * 2;                /* even k  */                  \
            int n4 = (f & 31) * 4;                                               \
            size_t ga = wbase + (size_t)(k0l + kk) * I_ + nbase + n4;            \
            float4 g0 = *(const float4*)(gate_w + ga);                           \
            float4 g1 = *(const float4*)(gate_w + ga + I_);                      \
            float4 u0 = *(const float4*)(up_w + ga);                             \
            float4 u1 = *(const float4*)(up_w + ga + I_);                        \
            if (it == 0 && k0l + 2 * KT < H_) {                                  \
                __builtin_prefetch(gate_w + ga + 2 * KT * (size_t)I_, 0, 0);     \
                __builtin_prefetch(up_w + ga + 2 * KT * (size_t)I_, 0, 0);       \
            }                                                                    \
            *(unsigned int*)&Bg[BUF][n4 + 0][kk] = pack2bf(g0.x, g1.x);          \
            *(unsigned int*)&Bg[BUF][n4 + 1][kk] = pack2bf(g0.y, g1.y);          \
            *(unsigned int*)&Bg[BUF][n4 + 2][kk] = pack2bf(g0.z, g1.z);          \
            *(unsigned int*)&Bg[BUF][n4 + 3][kk] = pack2bf(g0.w, g1.w);          \
            *(unsigned int*)&Bu[BUF][n4 + 0][kk] = pack2bf(u0.x, u1.x);          \
            *(unsigned int*)&Bu[BUF][n4 + 1][kk] = pack2bf(u0.y, u1.y);          \
            *(unsigned int*)&Bu[BUF][n4 + 2][kk] = pack2bf(u0.z, u1.z);          \
            *(unsigned int*)&Bu[BUF][n4 + 3][kk] = pack2bf(u0.w, u1.w);          \
        }                                                                        \
    }

    G1_LOAD(0, 0);
    for (int i = 0; i < NSTEPS; ++i) {
        wait_async_all();                      // ASYNCcnt drained (A tile in LDS)
        __syncthreads();                       // tile i ready in buffer i&1
        int cur = i & 1;
        if (i + 1 < NSTEPS) { G1_LOAD(cur ^ 1, (i + 1) * KT); }

        // distinct fragment registers -> partial dscnt waits, WMMA overlap
        FragBf a, bg[4], bu[4];
        a.q[0] = *(const uint4*)&As[cur][m_off + lm][laneHi * 16];
        a.q[1] = *(const uint4*)&As[cur][m_off + lm][laneHi * 16 + 8];
#pragma unroll
        for (int s = 0; s < 4; ++s) {
            int n = n_off + s * 16 + lm;
            bg[s].q[0] = *(const uint4*)&Bg[cur][n][laneHi * 16];
            bg[s].q[1] = *(const uint4*)&Bg[cur][n][laneHi * 16 + 8];
            bu[s].q[0] = *(const uint4*)&Bu[cur][n][laneHi * 16];
            bu[s].q[1] = *(const uint4*)&Bu[cur][n][laneHi * 16 + 8];
        }
#pragma unroll
        for (int s = 0; s < 4; ++s) {
            accG[s] = __builtin_amdgcn_wmma_f32_16x16x32_bf16(
                false, a.v, false, bg[s].v, (short)0, accG[s], false, false);
            accU[s] = __builtin_amdgcn_wmma_f32_16x16x32_bf16(
                false, a.v, false, bu[s].v, (short)0, accU[s], false, false);
        }
    }
#undef G1_LOAD

    // epilogue: silu(gate)*up -> bf16 activations (grouped layout)
#pragma unroll
    for (int s = 0; s < 4; ++s) {
#pragma unroll
        for (int r = 0; r < 8; ++r) {
            int m = m_off + r + laneHi * 8;
            if (m < mcount) {
                float g = accG[s][r], u = accU[s][r];
                float silu = g * __builtin_amdgcn_rcpf(1.f + __expf(-g));
                act[(size_t)(mbase + m) * I_ + nbase + n_off + s * 16 + lm] =
                    f2bf(silu * u);
            }
        }
    }
}

// ---------------------------------------------------------------------------
// 7) gemm2: out[t] += w * (act @ down_w[e])   [K-dim = I, N = H]
// ---------------------------------------------------------------------------
__global__ __launch_bounds__(256) void gemm2_kernel(
    const unsigned short* __restrict__ act,
    const float* __restrict__ down_w,
    const int* __restrict__ token_perm, const float* __restrict__ weight_perm,
    const int* __restrict__ offsets,
    const int* __restrict__ worklist, const int* __restrict__ total_tiles,
    float* __restrict__ out) {

    int tile = blockIdx.y;
    if (tile >= *total_tiles) return;
    int wle = worklist[tile];
    int e = wle >> 16, mt = wle & 0xFFFF;
    int gstart = offsets[e], gend = offsets[e + 1];
    int mbase  = gstart + mt * MT;
    int mcount = gend - mbase; if (mcount > MT) mcount = MT;
    int nbase  = blockIdx.x * NT;

    __shared__ __align__(16) unsigned short As[2][MT][LDP];
    __shared__ __align__(16) unsigned short Bs[2][NT][LDP];   // transposed [n][k]
    __shared__ int   s_tok[MT];
    __shared__ float s_w[MT];

    int tid = threadIdx.x;
    if (tid < MT) {
        int gr = mbase + tid;
        bool v = (tid < mcount);
        s_tok[tid] = v ? token_perm[gr]  : 0;
        s_w[tid]   = v ? weight_perm[gr] : 0.f;
    }

    int wid = tid >> 5, lane = tid & 31;
    int laneHi = lane >> 4, lm = lane & 15;
    int m_off = (wid & 3) * 16;
    int n_off = (wid >> 2) * 64;

    const int a_m = tid >> 2, a_q = tid & 3;
    const size_t wbase = (size_t)e * I_ * H_;

    v8f acc[4];
#pragma unroll
    for (int s = 0; s < 4; ++s)
        acc[s] = (v8f){0.f,0.f,0.f,0.f,0.f,0.f,0.f,0.f};

    constexpr int NSTEPS = I_ / KT;                           // 24
#define G2_LOAD(BUF, KK0)                                                        \
    {                                                                            \
        int k0l = (KK0);                                                         \
        async_copy_b128(&As[BUF][a_m][a_q * 8],                                  \
            (const uint4*)(act + (size_t)(mbase + a_m) * I_ + k0l) + a_q);       \
        _Pragma("unroll")                                                        \
        for (int it = 0; it < 2; ++it) {                                         \
            int f  = it * 256 + tid;                                             \
            int kk = (f >> 5) * 2;                                               \
            int n4 = (f & 31) * 4;                                               \
            size_t ga = wbase + (size_t)(k0l + kk) * H_ + nbase + n4;            \
            float4 d0 = *(const float4*)(down_w + ga);                           \
            float4 d1 = *(const float4*)(down_w + ga + H_);                      \
            if (it == 0 && k0l + 2 * KT < I_) {                                  \
                __builtin_prefetch(down_w + ga + 2 * KT * (size_t)H_, 0, 0);     \
            }                                                                    \
            *(unsigned int*)&Bs[BUF][n4 + 0][kk] = pack2bf(d0.x, d1.x);          \
            *(unsigned int*)&Bs[BUF][n4 + 1][kk] = pack2bf(d0.y, d1.y);          \
            *(unsigned int*)&Bs[BUF][n4 + 2][kk] = pack2bf(d0.z, d1.z);          \
            *(unsigned int*)&Bs[BUF][n4 + 3][kk] = pack2bf(d0.w, d1.w);          \
        }                                                                        \
    }

    __syncthreads();                            // s_tok/s_w visible
    G2_LOAD(0, 0);
    for (int i = 0; i < NSTEPS; ++i) {
        wait_async_all();
        __syncthreads();
        int cur = i & 1;
        if (i + 1 < NSTEPS) { G2_LOAD(cur ^ 1, (i + 1) * KT); }

        FragBf a, b[4];
        a.q[0] = *(const uint4*)&As[cur][m_off + lm][laneHi * 16];
        a.q[1] = *(const uint4*)&As[cur][m_off + lm][laneHi * 16 + 8];
#pragma unroll
        for (int s = 0; s < 4; ++s) {
            int n = n_off + s * 16 + lm;
            b[s].q[0] = *(const uint4*)&Bs[cur][n][laneHi * 16];
            b[s].q[1] = *(const uint4*)&Bs[cur][n][laneHi * 16 + 8];
        }
#pragma unroll
        for (int s = 0; s < 4; ++s) {
            acc[s] = __builtin_amdgcn_wmma_f32_16x16x32_bf16(
                false, a.v, false, b[s].v, (short)0, acc[s], false, false);
        }
    }
#undef G2_LOAD

    // epilogue: weighted atomic combine into out[t, :]
#pragma unroll
    for (int s = 0; s < 4; ++s) {
#pragma unroll
        for (int r = 0; r < 8; ++r) {
            int m = m_off + r + laneHi * 8;
            if (m < mcount) {
                float v = acc[s][r] * s_w[m];
                atomicAdd(&out[(size_t)s_tok[m] * H_ + nbase + n_off + s * 16 + lm], v);
            }
        }
    }
}

// ---------------------------------------------------------------------------
extern "C" void kernel_launch(void* const* d_in, const int* in_sizes, int n_in,
                              void* d_out, int out_size, void* d_ws, size_t ws_size,
                              hipStream_t stream) {
    const float* hs     = (const float*)d_in[0];
    const float* logits = (const float*)d_in[1];
    const float* gate_w = (const float*)d_in[2];
    const float* up_w   = (const float*)d_in[3];
    const float* down_w = (const float*)d_in[4];
    float* out = (float*)d_out;

    // carve workspace (256B-aligned chunks)
    char* p = (char*)d_ws;
    auto carve = [&](size_t bytes) -> void* {
        void* r = (void*)p;
        p += (bytes + 255) & ~(size_t)255;
        return r;
    };
    int*   counts      = (int*)carve(E_ * 4);
    int*   fill        = (int*)carve(E_ * 4);
    int*   offsets     = (int*)carve((E_ + 1) * 4);
    int*   total       = (int*)carve(4);
    int*   worklist    = (int*)carve(MAX_TILES * 4);
    int*   experts_sel = (int*)carve((size_t)T_ * K_ * 4);
    float* wts_sel     = (float*)carve((size_t)T_ * K_ * 4);
    int*   token_perm  = (int*)carve((size_t)T_ * K_ * 4);
    float* weight_perm = (float*)carve((size_t)T_ * K_ * 4);
    unsigned short* hs_bf = (unsigned short*)carve((size_t)T_ * H_ * 2);
    unsigned short* act   = (unsigned short*)carve(((size_t)T_ * K_ + MT) * I_ * 2);

    // 1) zero out + counters
    init_kernel<<<(T_ * H_ / 4) / 256, 256, 0, stream>>>(out, counts, fill, total);
    // 2) routing
    routing_kernel<<<(T_ + 255) / 256, 256, 0, stream>>>(logits, experts_sel, wts_sel, counts);
    // 3) scan + worklist
    scan_kernel<<<1, 64, 0, stream>>>(counts, offsets, worklist, total);
    // 4) scatter (counting sort)
    scatter_kernel<<<(T_ * K_ + 255) / 256, 256, 0, stream>>>(
        experts_sel, wts_sel, offsets, fill, token_perm, weight_perm);
    // 5) hs -> bf16
    convert_kernel<<<(T_ * H_ / 4) / 256, 256, 0, stream>>>(hs, hs_bf);
    // 6) gate/up grouped GEMM + swiglu
    {
        dim3 grid(I_ / NT, MAX_TILES);
        gemm1_kernel<<<grid, 256, 0, stream>>>(hs_bf, gate_w, up_w, token_perm,
                                               offsets, worklist, total, act);
    }
    // 7) down grouped GEMM + weighted combine
    {
        dim3 grid(H_ / NT, MAX_TILES);
        gemm2_kernel<<<grid, 256, 0, stream>>>(act, down_w, token_perm, weight_perm,
                                               offsets, worklist, total, out);
    }
}
